// GATEmbModel_74491912782414
// MI455X (gfx1250) — matrix operand order repaired and checked by reading, not compiled, not run on hardware
//
#include <hip/hip_runtime.h>
#include <hip/hip_bf16.h>

// ---------------------------------------------------------------------------
// 2-layer GAT + mean-pool + MLP scorer for MI455X (gfx1250, wave32, WMMA).
// FC projections run as v_wmma_f32_16x16x32_f16 (K=32 == HID in one WMMA).
// W is pre-shuffled into the per-lane B-fragment layout so the GEMM loads
// each fragment with one 32-byte vector load. Edge softmax uses ordered-uint
// atomicMax + atomicAdd passes (per-node state is L2-resident), message
// aggregation via global_atomic_add_f32.
// ---------------------------------------------------------------------------

#define HID   32
#define HEADS 4
#define HD    8
#define NEG_SLOPE 0.2f

typedef __attribute__((ext_vector_type(16))) _Float16 v16h;
typedef __attribute__((ext_vector_type(8)))  _Float16 v8h;
typedef __attribute__((ext_vector_type(8)))  float    v8f;
typedef __attribute__((ext_vector_type(4)))  float    f32x4;

// ---------- helpers --------------------------------------------------------

__device__ __forceinline__ unsigned f2ord(float f) {
    unsigned b = __float_as_uint(f);
    return (b & 0x80000000u) ? ~b : (b | 0x80000000u);
}
__device__ __forceinline__ float ord2f(unsigned u) {
    unsigned b = (u & 0x80000000u) ? (u & 0x7fffffffu) : ~u;
    return __uint_as_float(b);
}
__device__ __forceinline__ float lrelu(float x) {
    return x > 0.0f ? x : NEG_SLOPE * x;
}

// ---------- utility kernels ------------------------------------------------

__global__ void k_zero_f32(float* __restrict__ p, int n) {
    int t = blockIdx.x * blockDim.x + threadIdx.x;
    if (t < n) p[t] = 0.0f;
}

// Pre-shuffle W0/W1 (32x32 f32) into per-lane WMMA B-fragment layout:
//   Bpk[((L*2 + tile)*32 + lane)*16 + k] = W_L[(kb + k + (k>=8 ? 8 : 0))*32 + tile*16 + m]
// with m = lane&15, kb = (lane>>4)*8. Each lane's 16-half fragment is then
// one contiguous, 32B-aligned load in the GEMM.
__global__ void k_pack_w(const float* __restrict__ W0, const float* __restrict__ W1,
                         _Float16* __restrict__ Bpk) {
    int t = blockIdx.x * blockDim.x + threadIdx.x;
    if (t >= 2048) return;
    int k     = t & 15;
    int lane  = (t >> 4) & 31;
    int tileN = (t >> 9) & 1;
    int L     = t >> 10;
    int m  = lane & 15;
    int kb = (lane >> 4) * 8;
    const float* W = L ? W1 : W0;
    Bpk[t] = (_Float16)W[(kb + k + (k >= 8 ? 8 : 0)) * 32 + tileN * 16 + m];
}

// h16[n][c] = (f16) emb[node_ids[n]][c]
__global__ void k_gather_emb(const int* __restrict__ node_ids,
                             const float* __restrict__ emb,
                             _Float16* __restrict__ h16, int n32) {
    int t = blockIdx.x * blockDim.x + threadIdx.x;
    if (t >= n32) return;
    int n = t >> 5, c = t & 31;
    h16[t] = (_Float16)emb[(size_t)node_ids[n] * 32 + c];
}

// ---------- WMMA GEMM: F[N,32] = A16[N,32] @ W[32,32] ----------------------
// 128 threads = 4 waves; each wave owns one 16-row tile and issues 2 WMMAs
// (column halves) sharing the A fragment. EXEC is all-ones at the WMMAs.
__global__ void __launch_bounds__(128)
k_gemm_wmma(const _Float16* __restrict__ A16, const _Float16* __restrict__ Bpk,
            float* __restrict__ F, int nTiles, int nRows) {
    int wave = threadIdx.x >> 5;
    int lane = threadIdx.x & 31;
    int tile = blockIdx.x * 4 + wave;
    if (tile >= nTiles) return;                 // wave-uniform

    int m  = lane & 15;
    int hi = lane >> 4;

    int row = tile * 16 + m;
    if (row >= nRows) row = nRows - 1;          // clamp (select, EXEC intact)

    // A fragment (16-bit A 16x32 layout): two contiguous 16B chunks per lane.
    const _Float16* ap = A16 + (size_t)row * 32 + hi * 8;
    v8h alo = *(const v8h*)(ap);
    v8h ahi = *(const v8h*)(ap + 16);
    v16h a = __builtin_shufflevector(alo, ahi, 0, 1, 2, 3, 4, 5, 6, 7,
                                     8, 9, 10, 11, 12, 13, 14, 15);

    // B fragments: pre-shuffled, one 32B vector load each.
    v16h b0 = *(const v16h*)(Bpk + (size_t)lane * 16);
    v16h b1 = *(const v16h*)(Bpk + (size_t)(32 + lane) * 16);

    v8f c0 = {}, c1 = {};
    c0 = __builtin_amdgcn_wmma_f32_16x16x32_f16(false, a, false, b0,
                                                (short)0, c0, false, false);
    c1 = __builtin_amdgcn_wmma_f32_16x16x32_f16(false, a, false, b1,
                                                (short)0, c1, false, false);

    // D layout: VGPR r -> row (hi*8 + r), col = m (+16 for second tile).
    float* fp = F + (size_t)(tile * 16 + hi * 8) * 32 + m;
    if (tile * 16 + 16 <= nRows) {              // wave-uniform fast path
#pragma unroll
        for (int r = 0; r < 8; ++r) {
            fp[r * 32]      = c0[r];
            fp[r * 32 + 16] = c1[r];
        }
    } else {                                    // ragged tail (rare/never)
#pragma unroll
        for (int r = 0; r < 8; ++r) {
            if (tile * 16 + hi * 8 + r < nRows) {
                fp[r * 32]      = c0[r];
                fp[r * 32 + 16] = c1[r];
            }
        }
    }
}

// ---------- attention coefficients: el/er per (node, head) -----------------

__global__ void k_attn_coef(const float* __restrict__ F,
                            const float* __restrict__ al,
                            const float* __restrict__ ar,
                            float* __restrict__ el, float* __restrict__ er,
                            int n4) {
    int t = blockIdx.x * blockDim.x + threadIdx.x;
    if (t >= n4) return;
    int n = t >> 2, h = t & 3;
    const f32x4* fr = (const f32x4*)(F + (size_t)n * 32 + h * HD);
    const f32x4* alv = (const f32x4*)(al + h * HD);
    const f32x4* arv = (const f32x4*)(ar + h * HD);
    f32x4 f0 = fr[0], f1 = fr[1];
    f32x4 l0 = alv[0], l1 = alv[1];
    f32x4 r0 = arv[0], r1 = arv[1];
    float sl = 0.0f, sr = 0.0f;
#pragma unroll
    for (int d = 0; d < 4; ++d) {
        sl += f0[d] * l0[d] + f1[d] * l1[d];
        sr += f0[d] * r0[d] + f1[d] * r1[d];
    }
    el[t] = sl;
    er[t] = sr;
}

// ---------- edge pass 1: segment max (ordered-uint atomicMax) --------------

__global__ void k_edge_max(const int* __restrict__ src, const int* __restrict__ dst,
                           const float* __restrict__ el, const float* __restrict__ er,
                           unsigned* __restrict__ emaxu, int E) {
    int t = blockIdx.x * blockDim.x + threadIdx.x;
    if (t >= E) return;
    int s = src[t], d = dst[t];
    f32x4 elv = *(const f32x4*)(el + (size_t)s * 4);
    f32x4 erv = *(const f32x4*)(er + (size_t)d * 4);
#pragma unroll
    for (int h = 0; h < HEADS; ++h) {
        float e = lrelu(elv[h] + erv[h]);
        atomicMax(&emaxu[d * 4 + h], f2ord(e));
    }
}

// ---------- edge pass 2: softmax denominator -------------------------------

__global__ void k_edge_denom(const int* __restrict__ src, const int* __restrict__ dst,
                             const float* __restrict__ el, const float* __restrict__ er,
                             const unsigned* __restrict__ emaxu,
                             float* __restrict__ denom, int E) {
    int t = blockIdx.x * blockDim.x + threadIdx.x;
    if (t >= E) return;
    int s = src[t], d = dst[t];
    f32x4 elv = *(const f32x4*)(el + (size_t)s * 4);
    f32x4 erv = *(const f32x4*)(er + (size_t)d * 4);
    const unsigned* mu = emaxu + (size_t)d * 4;
#pragma unroll
    for (int h = 0; h < HEADS; ++h) {
        float e  = lrelu(elv[h] + erv[h]);
        float ex = __expf(e - ord2f(mu[h]));
        atomicAdd(&denom[d * 4 + h], ex);
    }
}

// ---------- edge pass 3: weighted message scatter --------------------------
// one thread per (edge, head): 2x b128 gathers + 8 f32 atomic adds.

__global__ void k_edge_msg(const int* __restrict__ src, const int* __restrict__ dst,
                           const float* __restrict__ el, const float* __restrict__ er,
                           const unsigned* __restrict__ emaxu,
                           const float* __restrict__ denom,
                           const float* __restrict__ F,
                           float* __restrict__ acc, int E4) {
    int t = blockIdx.x * blockDim.x + threadIdx.x;
    if (t >= E4) return;
    int eid = t >> 2, h = t & 3;
    int s = src[eid], d = dst[eid];
    float e     = lrelu(el[s * 4 + h] + er[d * 4 + h]);
    float ex    = __expf(e - ord2f(emaxu[d * 4 + h]));
    float alpha = ex / denom[d * 4 + h];
    const f32x4* fr = (const f32x4*)(F + (size_t)s * 32 + h * HD);
    f32x4 f0 = fr[0], f1 = fr[1];
    float* ob = acc + (size_t)d * 32 + h * HD;
#pragma unroll
    for (int k = 0; k < 4; ++k) {
        atomicAdd(&ob[k],     alpha * f0[k]);
        atomicAdd(&ob[4 + k], alpha * f1[k]);
    }
}

// ---------- epilogues ------------------------------------------------------

// layer 0: bias + relu, re-emit as f16 for next WMMA GEMM
__global__ void k_epilogue0(const float* __restrict__ acc, const float* __restrict__ b,
                            _Float16* __restrict__ h16, int n32) {
    int t = blockIdx.x * blockDim.x + threadIdx.x;
    if (t >= n32) return;
    int c = t & 31;
    float v = acc[t] + b[c];
    h16[t] = (_Float16)(v > 0.0f ? v : 0.0f);
}

// layer 1: bias + relu, fused per-graph sum pooling
__global__ void k_epilogue1(const float* __restrict__ acc, const float* __restrict__ b,
                            const int* __restrict__ graph_ids,
                            float* __restrict__ gsum, float* __restrict__ gcnt,
                            int n32) {
    int t = blockIdx.x * blockDim.x + threadIdx.x;
    if (t >= n32) return;
    int n = t >> 5, c = t & 31;
    float v = acc[t] + b[c];
    v = v > 0.0f ? v : 0.0f;
    int g = graph_ids[n];
    atomicAdd(&gsum[g * 32 + c], v);
    if (c == 0) atomicAdd(&gcnt[g], 1.0f);
}

// ---------- scorer MLP: relu(hg@sw1+sb1)@sw2+sb2 ---------------------------

__global__ void k_scorer(const float* __restrict__ gsum, const float* __restrict__ gcnt,
                         const float* __restrict__ sw1, const float* __restrict__ sb1,
                         const float* __restrict__ sw2, const float* __restrict__ sb2,
                         float* __restrict__ out, int Bn) {
    int g = blockIdx.x * blockDim.x + threadIdx.x;
    if (g >= Bn) return;
    float cnt = gcnt[g];
    if (cnt < 1.0f) cnt = 1.0f;
    float hg[HID];
#pragma unroll
    for (int c = 0; c < HID; ++c) hg[c] = gsum[g * 32 + c] / cnt;
    float o = sb2[0];
    for (int j = 0; j < HID; ++j) {
        float sacc = sb1[j];
#pragma unroll
        for (int i = 0; i < HID; ++i) sacc += hg[i] * sw1[i * 32 + j];
        if (sacc < 0.0f) sacc = 0.0f;
        o += sacc * sw2[j];
    }
    out[g] = o;
}

// ---------------------------------------------------------------------------

extern "C" void kernel_launch(void* const* d_in, const int* in_sizes, int n_in,
                              void* d_out, int out_size, void* d_ws, size_t ws_size,
                              hipStream_t stream) {
    const int*   node_ids  = (const int*)  d_in[0];
    const int*   src       = (const int*)  d_in[1];
    const int*   dst       = (const int*)  d_in[2];
    const int*   graph_ids = (const int*)  d_in[3];
    const float* emb       = (const float*)d_in[4];
    const float* W0        = (const float*)d_in[5];
    const float* al0       = (const float*)d_in[6];
    const float* ar0       = (const float*)d_in[7];
    const float* b0        = (const float*)d_in[8];
    const float* W1        = (const float*)d_in[9];
    const float* al1       = (const float*)d_in[10];
    const float* ar1       = (const float*)d_in[11];
    const float* b1        = (const float*)d_in[12];
    const float* sw1       = (const float*)d_in[13];
    const float* sb1       = (const float*)d_in[14];
    const float* sw2       = (const float*)d_in[15];
    const float* sb2       = (const float*)d_in[16];
    float* out = (float*)d_out;

    const int N  = in_sizes[0];
    const int E  = in_sizes[1];
    const int Bn = out_size;            // [B,1] -> B graphs

    // ---- workspace layout (bytes) ----
    char* w = (char*)d_ws;
    size_t off = 0;
    float*    F      = (float*)   (w + off); off += (size_t)N * 32 * 4;
    // acc, emaxu, denom contiguous -> single zero pass over N*40 floats
    float*    acc    = (float*)   (w + off); off += (size_t)N * 32 * 4;
    unsigned* emaxu  = (unsigned*)(w + off); off += (size_t)N * 4 * 4;
    float*    denom  = (float*)   (w + off); off += (size_t)N * 4 * 4;
    float*    el     = (float*)   (w + off); off += (size_t)N * 4 * 4;
    float*    er     = (float*)   (w + off); off += (size_t)N * 4 * 4;
    _Float16* h16    = (_Float16*)(w + off); off += (size_t)N * 32 * 2;
    _Float16* Bpk    = (_Float16*)(w + off); off += 2048 * 2;   // B-fragment W
    float*    gsum   = (float*)   (w + off); off += (size_t)Bn * 32 * 4;
    float*    gcnt   = (float*)   (w + off); off += (size_t)Bn * 4;

    const int TB = 256;
    auto cdiv = [](int a, int b) { return (a + b - 1) / b; };

    // ---- prep ----
    k_pack_w<<<cdiv(2048, TB), TB, 0, stream>>>(W0, W1, Bpk);
    k_gather_emb<<<cdiv(N * 32, TB), TB, 0, stream>>>(node_ids, emb, h16, N * 32);
    k_zero_f32<<<cdiv(Bn * 33, TB), TB, 0, stream>>>(gsum, Bn * 33); // gsum + gcnt

    const int nTiles = (N + 15) / 16;

    for (int L = 0; L < 2; ++L) {
        const _Float16* Wl = Bpk + (size_t)L * 1024;
        const float* al = (L == 0) ? al0 : al1;
        const float* ar = (L == 0) ? ar0 : ar1;
        const float* bb = (L == 0) ? b0  : b1;

        // zero acc + emaxu + denom (contiguous N*40 floats)
        k_zero_f32<<<cdiv(N * 40, TB), TB, 0, stream>>>(acc, N * 40);

        // FC projection via WMMA
        k_gemm_wmma<<<cdiv(nTiles, 4), 128, 0, stream>>>(h16, Wl, F, nTiles, N);

        // per-(node,head) attention coefficients
        k_attn_coef<<<cdiv(N * 4, TB), TB, 0, stream>>>(F, al, ar, el, er, N * 4);

        // edge softmax + aggregation
        k_edge_max  <<<cdiv(E, TB), TB, 0, stream>>>(src, dst, el, er, emaxu, E);
        k_edge_denom<<<cdiv(E, TB), TB, 0, stream>>>(src, dst, el, er, emaxu, denom, E);
        k_edge_msg  <<<cdiv(E * 4, TB), TB, 0, stream>>>(src, dst, el, er, emaxu,
                                                         denom, F, acc, E * 4);

        if (L == 0)
            k_epilogue0<<<cdiv(N * 32, TB), TB, 0, stream>>>(acc, bb, h16, N * 32);
        else
            k_epilogue1<<<cdiv(N * 32, TB), TB, 0, stream>>>(acc, bb, graph_ids,
                                                             gsum, gcnt, N * 32);
    }

    // scorer MLP over pooled graph features
    k_scorer<<<cdiv(Bn, 64), 64, 0, stream>>>(gsum, gcnt, sw1, sb1, sw2, sb2, out, Bn);
}
